// MotionCompleteNet_62929860821095
// MI455X (gfx1250) — matrix operand order repaired and checked by reading, not compiled, not run on hardware
//
#include <hip/hip_runtime.h>
#include <hip/hip_bf16.h>
#include <cmath>

// ---------------------------------------------------------------------------
// MotionCompleteNet for MI455X (gfx1250, wave32, WMMA).
// GEMMs (q/k/v/skip linears) use v_wmma_f32_16x16x32_f16: f16 inputs, f32
// accumulate. Everything else (LN, attention softmax, scatter) stays f32.
// ---------------------------------------------------------------------------

typedef __attribute__((ext_vector_type(16))) _Float16 v16h;
typedef __attribute__((ext_vector_type(8)))  _Float16 v8h;
typedef __attribute__((ext_vector_type(8)))  float    v8f;

static constexpr int cN0 = 65536, cN1 = 32768, cN2 = 16384, cN3 = 8192;
static constexpr int cE0 = 524288, cE1 = 262144, cE2 = 131072, cE3 = 65536;
static constexpr int cT  = 16;

// ------------------------------- utilities ---------------------------------

__device__ __forceinline__ float sigmoidf_(float x) { return 1.0f / (1.0f + __expf(-x)); }

__device__ __forceinline__ void atomicMaxF(float* addr, float val) {
  unsigned* ua = (unsigned*)addr;
  unsigned cur = __float_as_uint(*addr);
  while (__uint_as_float(cur) < val) {
    unsigned prev = atomicCAS(ua, cur, __float_as_uint(val));
    if (prev == cur) break;
    cur = prev;
  }
}

// ------------------------- fused LSTM + seq head + encoder -----------------
// One wave per node; lane L owns hidden unit L of both LSTM layers.
// Weights staged in LDS *transposed* ([k][128]) so per-k gate FMAs are
// bank-conflict free across lanes.

static constexpr int S_WIH1 = 0;              // [4][128]
static constexpr int S_WHH1 = 512;            // [32][128]
static constexpr int S_B1   = 4608;           // [128] (bih+bhh)
static constexpr int S_WIH2 = 4736;           // [32][128]
static constexpr int S_WHH2 = 8832;           // [32][128]
static constexpr int S_B2   = 12928;          // [128]
static constexpr int S_WSL  = 13056;          // [32][4]
static constexpr int S_BSL  = 13184;          // [4]
static constexpr int S_WENC = 13188;          // [11][32]
static constexpr int S_BENC = 13540;          // [32]
static constexpr int S_TOT  = 13572;

__global__ __launch_bounds__(256) void k_lstm_enc(
    const float* __restrict__ prev,   // [T, N0, 4]
    const float* __restrict__ cpos,   // [N0, 3]
    const float* __restrict__ cmot,   // [N0, 4]
    const float* __restrict__ Wih1, const float* __restrict__ Whh1,
    const float* __restrict__ bih1, const float* __restrict__ bhh1,
    const float* __restrict__ Wih2, const float* __restrict__ Whh2,
    const float* __restrict__ bih2, const float* __restrict__ bhh2,
    const float* __restrict__ Wsl,  const float* __restrict__ bsl,
    const float* __restrict__ Wenc, const float* __restrict__ benc,
    float* __restrict__ xout)         // [N0, 32]
{
  __shared__ float sm[S_TOT];
  for (int i = threadIdx.x; i < 512;  i += 256) { int gi = i >> 2, k = i & 3;  sm[S_WIH1 + k*128 + gi] = Wih1[i]; }
  for (int i = threadIdx.x; i < 4096; i += 256) { int gi = i >> 5, k = i & 31; sm[S_WHH1 + k*128 + gi] = Whh1[i]; }
  for (int i = threadIdx.x; i < 128;  i += 256) sm[S_B1 + i] = bih1[i] + bhh1[i];
  for (int i = threadIdx.x; i < 4096; i += 256) { int gi = i >> 5, k = i & 31; sm[S_WIH2 + k*128 + gi] = Wih2[i]; }
  for (int i = threadIdx.x; i < 4096; i += 256) { int gi = i >> 5, k = i & 31; sm[S_WHH2 + k*128 + gi] = Whh2[i]; }
  for (int i = threadIdx.x; i < 128;  i += 256) sm[S_B2 + i] = bih2[i] + bhh2[i];
  for (int i = threadIdx.x; i < 128;  i += 256) sm[S_WSL + i] = Wsl[i];
  if (threadIdx.x < 4)  sm[S_BSL + threadIdx.x] = bsl[threadIdx.x];
  for (int i = threadIdx.x; i < 352;  i += 256) sm[S_WENC + i] = Wenc[i];
  if (threadIdx.x < 32) sm[S_BENC + threadIdx.x] = benc[threadIdx.x];
  __syncthreads();

  const int wave = threadIdx.x >> 5, lane = threadIdx.x & 31;
  const int node = blockIdx.x * 8 + wave;

  float h1 = 0.f, c1 = 0.f, h2 = 0.f, c2 = 0.f;
  const int g0 = lane, g1 = 32 + lane, g2 = 64 + lane, g3 = 96 + lane;

  for (int t = 0; t < cT; ++t) {
    const float* pm = prev + (((size_t)t * cN0 + node) << 2);
    const float x0 = pm[0], x1 = pm[1], x2 = pm[2], x3 = pm[3];

    // ---- layer 1 ----
    float p0 = sm[S_B1+g0], p1 = sm[S_B1+g1], p2 = sm[S_B1+g2], p3 = sm[S_B1+g3];
    {
      const float xs[4] = {x0, x1, x2, x3};
      #pragma unroll
      for (int k = 0; k < 4; ++k) {
        const float xv = xs[k];
        p0 += sm[S_WIH1 + k*128 + g0] * xv; p1 += sm[S_WIH1 + k*128 + g1] * xv;
        p2 += sm[S_WIH1 + k*128 + g2] * xv; p3 += sm[S_WIH1 + k*128 + g3] * xv;
      }
      #pragma unroll
      for (int k = 0; k < 32; ++k) {
        const float hk = __shfl(h1, k, 32);
        p0 += sm[S_WHH1 + k*128 + g0] * hk; p1 += sm[S_WHH1 + k*128 + g1] * hk;
        p2 += sm[S_WHH1 + k*128 + g2] * hk; p3 += sm[S_WHH1 + k*128 + g3] * hk;
      }
      const float ig = sigmoidf_(p0), fg = sigmoidf_(p1), gg = tanhf(p2), og = sigmoidf_(p3);
      c1 = fg * c1 + ig * gg;
      h1 = og * tanhf(c1);
    }
    // ---- layer 2 (input = h1 of this step) ----
    p0 = sm[S_B2+g0]; p1 = sm[S_B2+g1]; p2 = sm[S_B2+g2]; p3 = sm[S_B2+g3];
    #pragma unroll
    for (int k = 0; k < 32; ++k) {
      const float ak = __shfl(h1, k, 32);
      const float bk = __shfl(h2, k, 32);
      p0 += sm[S_WIH2 + k*128 + g0] * ak + sm[S_WHH2 + k*128 + g0] * bk;
      p1 += sm[S_WIH2 + k*128 + g1] * ak + sm[S_WHH2 + k*128 + g1] * bk;
      p2 += sm[S_WIH2 + k*128 + g2] * ak + sm[S_WHH2 + k*128 + g2] * bk;
      p3 += sm[S_WIH2 + k*128 + g3] * ak + sm[S_WHH2 + k*128 + g3] * bk;
    }
    const float ig = sigmoidf_(p0), fg = sigmoidf_(p1), gg = tanhf(p2), og = sigmoidf_(p3);
    c2 = fg * c2 + ig * gg;
    h2 = og * tanhf(c2);
  }

  // ---- seq_lin head: seq_pred[j] = bsl[j] + sum_L h2_L * Wsl[L,j] ----
  float s0 = h2 * sm[S_WSL + lane*4 + 0];
  float s1 = h2 * sm[S_WSL + lane*4 + 1];
  float s2 = h2 * sm[S_WSL + lane*4 + 2];
  float s3 = h2 * sm[S_WSL + lane*4 + 3];
  #pragma unroll
  for (int m = 16; m; m >>= 1) {
    s0 += __shfl_xor(s0, m, 32); s1 += __shfl_xor(s1, m, 32);
    s2 += __shfl_xor(s2, m, 32); s3 += __shfl_xor(s3, m, 32);
  }
  s0 += sm[S_BSL+0]; s1 += sm[S_BSL+1]; s2 += sm[S_BSL+2]; s3 += sm[S_BSL+3];

  // ---- encoder: x = concat(curr_pos, seq_pred, curr_motion) @ Wenc + benc ----
  float in[11];
  in[0] = cpos[(size_t)node*3 + 0]; in[1] = cpos[(size_t)node*3 + 1]; in[2] = cpos[(size_t)node*3 + 2];
  in[3] = s0; in[4] = s1; in[5] = s2; in[6] = s3;
  in[7] = cmot[(size_t)node*4 + 0]; in[8] = cmot[(size_t)node*4 + 1];
  in[9] = cmot[(size_t)node*4 + 2]; in[10] = cmot[(size_t)node*4 + 3];
  float xv = sm[S_BENC + lane];
  #pragma unroll
  for (int k = 0; k < 11; ++k) xv += in[k] * sm[S_WENC + k*32 + lane];
  xout[(size_t)node * 32 + lane] = xv;
}

// -------------------- weight packing into WMMA B fragments -----------------
// B (32x16 f16 per k-step tile): lane l holds column n = l&15, halves h=0..15
// map to K = 16*(l>>4) + h (contiguous k order, chosen here; A uses the same).

__global__ void k_pack4(const float* __restrict__ Wq, const float* __restrict__ Wk,
                        const float* __restrict__ Wv, const float* __restrict__ Ws,
                        _Float16* __restrict__ dst, int C)
{
  const int total = C * C;
  const int t = blockIdx.x * blockDim.x + threadIdx.x;
  if (t >= 4 * total) return;
  const int mat = t / total, r = t - mat * total;
  const float* W = (mat == 0) ? Wq : (mat == 1) ? Wk : (mat == 2) ? Wv : Ws;
  const int ktiles = C >> 5;
  const int h    = r & 15;
  const int lane = (r >> 4) & 31;
  const int tile = r >> 9;
  const int kt = tile % ktiles, ntile = tile / ktiles;
  const int n = ntile * 16 + (lane & 15);
  const int k = kt * 32 + ((lane >> 4) << 4) + h;
  dst[(size_t)mat * total + r] = (_Float16)W[(size_t)k * C + n];   // W is [din, dout]
}

// ----------------------- LN + ReLU -> f16 activations ----------------------

__global__ __launch_bounds__(256) void k_ln_relu_f16(
    const float* __restrict__ x, const float* __restrict__ g, const float* __restrict__ be,
    _Float16* __restrict__ out, int N, int C, int doLN)
{
  const int wave = threadIdx.x >> 5, lane = threadIdx.x & 31;
  const int node = blockIdx.x * 8 + wave;
  if (node >= N) return;
  const int cpl = C >> 5;
  const float* row = x + (size_t)node * C;
  float vals[4];
  #pragma unroll
  for (int r = 0; r < 4; ++r) if (r < cpl) vals[r] = row[lane + 32 * r];
  if (doLN) {
    float s = 0.f;
    #pragma unroll
    for (int r = 0; r < 4; ++r) if (r < cpl) s += vals[r];
    #pragma unroll
    for (int m = 16; m; m >>= 1) s += __shfl_xor(s, m, 32);
    const float mean = s / (float)C;
    float v2 = 0.f;
    #pragma unroll
    for (int r = 0; r < 4; ++r) if (r < cpl) { const float d = vals[r] - mean; v2 += d * d; }
    #pragma unroll
    for (int m = 16; m; m >>= 1) v2 += __shfl_xor(v2, m, 32);
    const float rstd = rsqrtf(v2 / (float)C + 1e-5f);
    #pragma unroll
    for (int r = 0; r < 4; ++r) if (r < cpl) {
      const int ch = lane + 32 * r;
      const float y = (vals[r] - mean) * rstd * g[ch] + be[ch];
      vals[r] = fmaxf(y, 0.f);
    }
  }
  #pragma unroll
  for (int r = 0; r < 4; ++r) if (r < cpl) out[(size_t)node * C + lane + 32 * r] = (_Float16)vals[r];
}

// --------------------------- WMMA q/k/v/skip GEMM --------------------------
// One wave per 16x16 output tile; grid.z selects the matrix (0=q,1=k,2=v,3=s).
// A fragment: lane m = l&15, kbase = 16*(l>>4);
// halves 0..15 <- K[kbase..kbase+15] (matches packed B order).

__global__ __launch_bounds__(256) void k_wmma_qkvs(
    const _Float16* __restrict__ h16, const _Float16* __restrict__ pw,
    const float* __restrict__ bq, const float* __restrict__ bk,
    const float* __restrict__ bv, const float* __restrict__ bs,
    float* __restrict__ q, float* __restrict__ k, float* __restrict__ v, float* __restrict__ s,
    int N, int C)
{
  const int wave = threadIdx.x >> 5, lane = threadIdx.x & 31;
  const int mtile = blockIdx.x * 8 + wave;
  const int ntile = blockIdx.y;
  const int mat = blockIdx.z;
  if (mtile * 16 >= N) return;

  const _Float16* pmat = pw + (size_t)mat * C * C;
  const float* bias = (mat == 0) ? bq : (mat == 1) ? bk : (mat == 2) ? bv : bs;
  float* dst = (mat == 0) ? q : (mat == 1) ? k : (mat == 2) ? v : s;

  const int ktiles = C >> 5;
  const int m = lane & 15;
  const int kb = (lane >> 4) << 4;
  const _Float16* arow = h16 + (size_t)(mtile * 16 + m) * C;
  const _Float16* brow = pmat + (size_t)(ntile * ktiles) * 512 + lane * 16;

  v8f acc = {};
  for (int kt = 0; kt < ktiles; ++kt) {
    const v8h lo = *(const v8h*)(arow + kt * 32 + kb);
    const v8h hi = *(const v8h*)(arow + kt * 32 + kb + 8);
    const v16h A = __builtin_shufflevector(lo, hi, 0,1,2,3,4,5,6,7,8,9,10,11,12,13,14,15);
    const v16h B = *(const v16h*)(brow + kt * 512);
    acc = __builtin_amdgcn_wmma_f32_16x16x32_f16(false, A, false, B, (short)0, acc, false, false);
  }

  const int n = ntile * 16 + (lane & 15);
  const float bb = bias[n];
  const int mbase = mtile * 16 + ((lane >> 4) << 3);   // D: VGPR r -> M = r + 8*(lane>>4)
  #pragma unroll
  for (int r = 0; r < 8; ++r)
    dst[(size_t)(mbase + r) * C + n] = acc[r] + bb;
}

// ----------------------------- edge attention ------------------------------

__global__ void k_init_seg(float* __restrict__ amax, float* __restrict__ den, int n) {
  const int i = blockIdx.x * blockDim.x + threadIdx.x;
  if (i < n) { amax[i] = -3.0e38f; den[i] = 0.f; }
}

__global__ void k_edge_dot_max(const int* __restrict__ ei, int E,
                               const float* __restrict__ q, const float* __restrict__ k,
                               float* __restrict__ a, float* __restrict__ amax,
                               int C, float scale)
{
  const int e = blockIdx.x * blockDim.x + threadIdx.x;
  if (e >= E) return;
  const int s = ei[e], d = ei[E + e];
  const float4* qr = (const float4*)(q + (size_t)d * C);
  const float4* kr = (const float4*)(k + (size_t)s * C);
  float acc = 0.f;
  const int quads = C >> 2;
  for (int c = 0; c < quads; ++c) {
    const float4 qq = qr[c], kk = kr[c];
    acc += qq.x * kk.x + qq.y * kk.y + qq.z * kk.z + qq.w * kk.w;
  }
  acc *= scale;
  a[e] = acc;
  atomicMaxF(&amax[d], acc);
}

__global__ void k_edge_exp_sum(const int* __restrict__ ei, int E,
                               float* __restrict__ a, const float* __restrict__ amax,
                               float* __restrict__ den)
{
  const int e = blockIdx.x * blockDim.x + threadIdx.x;
  if (e >= E) return;
  const int d = ei[E + e];
  const float w = expf(a[e] - amax[d]);
  a[e] = w;
  atomicAdd(&den[d], w);
}

__global__ void k_edge_scatter(const int* __restrict__ ei, int E,
                               const float* __restrict__ a, const float* __restrict__ den,
                               const float* __restrict__ v, float* __restrict__ agg, int C)
{
  const unsigned quads = (unsigned)(C >> 2);
  const unsigned t = blockIdx.x * blockDim.x + threadIdx.x;
  if (t >= (unsigned)E * quads) return;
  const int e = (int)(t / quads);
  const int cq = (int)(t - (unsigned)e * quads);
  const int s = ei[e], d = ei[E + e];
  const float w = a[e] / den[d];
  const float4 vv = *(const float4*)(v + (size_t)s * C + cq * 4);
  float* o = agg + (size_t)d * C + cq * 4;
  atomicAdd(o + 0, w * vv.x);
  atomicAdd(o + 1, w * vv.y);
  atomicAdd(o + 2, w * vv.z);
  atomicAdd(o + 3, w * vv.w);
}

// --------------------------- elementwise plumbing --------------------------

__global__ void k_add_inplace(float* __restrict__ x, const float* __restrict__ a, unsigned n) {
  const unsigned i = blockIdx.x * blockDim.x + threadIdx.x;
  if (i < n) x[i] += a[i];
}

__global__ void k_gather(float* __restrict__ out, const float* __restrict__ in,
                         const int* __restrict__ idx, int N, int C) {
  const unsigned t = blockIdx.x * blockDim.x + threadIdx.x;
  if (t >= (unsigned)N * (unsigned)C) return;
  const int i = (int)(t / (unsigned)C), c = (int)(t % (unsigned)C);
  out[t] = in[(size_t)idx[i] * C + c];
}

__global__ void k_concat(float* __restrict__ out,
                         const float* __restrict__ a, const int* __restrict__ idx, int C1,
                         const float* __restrict__ b, int C2, int N) {
  const int Cc = C1 + C2;
  const unsigned t = blockIdx.x * blockDim.x + threadIdx.x;
  if (t >= (unsigned)N * (unsigned)Cc) return;
  const int i = (int)(t / (unsigned)Cc), c = (int)(t % (unsigned)Cc);
  out[t] = (c < C1) ? a[(size_t)idx[i] * C1 + c] : b[(size_t)i * C2 + (c - C1)];
}

// ------------------------------- output head -------------------------------

__global__ __launch_bounds__(256) void k_final_head(
    const float* __restrict__ f7, const float* __restrict__ g, const float* __restrict__ be,
    const float* __restrict__ Wlin, const float* __restrict__ blin,
    float* __restrict__ out, int N)
{
  const int wave = threadIdx.x >> 5, lane = threadIdx.x & 31;
  const int node = blockIdx.x * 8 + wave;
  if (node >= N) return;
  const float* row = f7 + (size_t)node * 128;
  float vals[4];
  #pragma unroll
  for (int r = 0; r < 4; ++r) vals[r] = row[lane + 32 * r];
  float s = vals[0] + vals[1] + vals[2] + vals[3];
  #pragma unroll
  for (int m = 16; m; m >>= 1) s += __shfl_xor(s, m, 32);
  const float mean = s * (1.f / 128.f);
  float v2 = 0.f;
  #pragma unroll
  for (int r = 0; r < 4; ++r) { const float d = vals[r] - mean; v2 += d * d; }
  #pragma unroll
  for (int m = 16; m; m >>= 1) v2 += __shfl_xor(v2, m, 32);
  const float rstd = rsqrtf(v2 * (1.f / 128.f) + 1e-5f);

  float p0 = 0.f, p1 = 0.f, p2 = 0.f, p3 = 0.f;
  #pragma unroll
  for (int r = 0; r < 4; ++r) {
    const int ch = lane + 32 * r;
    const float y = fmaxf((vals[r] - mean) * rstd * g[ch] + be[ch], 0.f);
    p0 += y * Wlin[ch * 4 + 0]; p1 += y * Wlin[ch * 4 + 1];
    p2 += y * Wlin[ch * 4 + 2]; p3 += y * Wlin[ch * 4 + 3];
  }
  #pragma unroll
  for (int m = 16; m; m >>= 1) {
    p0 += __shfl_xor(p0, m, 32); p1 += __shfl_xor(p1, m, 32);
    p2 += __shfl_xor(p2, m, 32); p3 += __shfl_xor(p3, m, 32);
  }
  if (lane == 0) {
    out[(size_t)node * 4 + 0] = p0 + blin[0];
    out[(size_t)node * 4 + 1] = p1 + blin[1];
    out[(size_t)node * 4 + 2] = p2 + blin[2];
    const float z = p3 + blin[3];
    out[(size_t)node * 4 + 3] = (z > 20.f) ? z : log1pf(expf(z));   // softplus
  }
}

// ------------------------------- host driver -------------------------------

extern "C" void kernel_launch(void* const* d_in, const int* in_sizes, int n_in,
                              void* d_out, int out_size, void* d_ws, size_t ws_size,
                              hipStream_t stream) {
  (void)in_sizes; (void)n_in; (void)out_size; (void)ws_size;
  auto F = [&](int i) { return (const float*)d_in[i]; };
  auto I = [&](int i) { return (const int*)d_in[i]; };

  // ---- workspace bump allocator ----
  char* wsp = (char*)d_ws;
  size_t off = 0;
  auto alloc = [&](size_t bytes) -> char* {
    off = (off + 255) & ~(size_t)255;
    char* p = wsp + off;
    off += bytes;
    return p;
  };
  _Float16* pw   = (_Float16*)alloc((size_t)4 * 128 * 128 * 2);
  _Float16* h16  = (_Float16*)alloc((size_t)cN0 * 128 * 2);
  float* qb   = (float*)alloc((size_t)cN0 * 128 * 4);
  float* kb   = (float*)alloc((size_t)cN0 * 128 * 4);
  float* vb   = (float*)alloc((size_t)cN0 * 128 * 4);
  float* aggb = (float*)alloc((size_t)cN0 * 128 * 4);
  float* ab    = (float*)alloc((size_t)cE0 * 4);
  float* amaxb = (float*)alloc((size_t)cN0 * 4);
  float* denb  = (float*)alloc((size_t)cN0 * 4);
  float* xb = (float*)alloc((size_t)cN0 * 32 * 4);
  float* f1 = (float*)alloc((size_t)cN0 * 32 * 4);
  float* f2 = (float*)alloc((size_t)cN1 * 32 * 4);
  float* f3 = (float*)alloc((size_t)cN2 * 32 * 4);
  float* f4 = (float*)alloc((size_t)cN3 * 32 * 4);
  float* f5 = (float*)alloc((size_t)cN2 * 64 * 4);
  float* f6 = (float*)alloc((size_t)cN1 * 96 * 4);
  float* f7 = (float*)alloc((size_t)cN0 * 128 * 4);

  const int* e0 = I(3); const int* e1 = I(4); const int* e2 = I(5); const int* e3 = I(6);
  const int* dn0 = I(7); const int* dn1 = I(8); const int* dn2 = I(9);
  const int* up0 = I(10); const int* up1 = I(11); const int* up2 = I(12);

  // ---- stage 1: LSTM x2 + seq head + encoder, fused ----
  k_lstm_enc<<<dim3(cN0 / 8), 256, 0, stream>>>(
      F(2), F(0), F(1),
      F(15), F(16), F(17), F(18),     // lstm[0]: Wih, Whh, bih, bhh
      F(19), F(20), F(21), F(22),     // lstm[1]
      F(23), F(24),                   // seq_lin W, b
      F(13), F(14),                   // enc W, b
      xb);

  // ---- TransformerConv building block ----
  auto tconv = [&](float* x, int N, int C, const int* ei, int E, int pqw,
                   const float* g, const float* be, float* copyOut) {
    const int total = 4 * C * C;
    k_pack4<<<dim3((total + 255) / 256), 256, 0, stream>>>(F(pqw), F(pqw + 2), F(pqw + 4), F(pqw + 6), pw, C);
    k_ln_relu_f16<<<dim3(N / 8), 256, 0, stream>>>(x, g, be, h16, N, C, g ? 1 : 0);
    k_wmma_qkvs<<<dim3(N / 128, C / 16, 4), 256, 0, stream>>>(
        h16, pw, F(pqw + 1), F(pqw + 3), F(pqw + 5), F(pqw + 7), qb, kb, vb, aggb, N, C);
    k_init_seg<<<dim3((N + 255) / 256), 256, 0, stream>>>(amaxb, denb, N);
    k_edge_dot_max<<<dim3((E + 255) / 256), 256, 0, stream>>>(ei, E, qb, kb, ab, amaxb, C, 1.0f / sqrtf((float)C));
    k_edge_exp_sum<<<dim3((E + 255) / 256), 256, 0, stream>>>(ei, E, ab, amaxb, denb);
    const unsigned work = (unsigned)E * (unsigned)(C >> 2);
    k_edge_scatter<<<dim3((work + 255) / 256), 256, 0, stream>>>(ei, E, ab, denb, vb, aggb, C);
    if (copyOut)
      hipMemcpyAsync(copyOut, aggb, (size_t)N * C * sizeof(float), hipMemcpyDeviceToDevice, stream);
    else {
      const unsigned n = (unsigned)N * (unsigned)C;
      k_add_inplace<<<dim3((n + 255) / 256), 256, 0, stream>>>(x, aggb, n);
    }
  };
  auto dlayer = [&](float* x, int N, int C, const int* ei, int E, int li) {
    const int base = 33 + 10 * li;   // g, be, qW, qb, kW, kb, vW, vb, sW, sb
    tconv(x, N, C, ei, E, base + 2, F(base), F(base + 1), nullptr);
  };
  auto gather = [&](float* out, const float* in, const int* idx, int N, int C) {
    const unsigned n = (unsigned)N * (unsigned)C;
    k_gather<<<dim3((n + 255) / 256), 256, 0, stream>>>(out, in, idx, N, C);
  };
  auto concat = [&](float* out, const float* a, const int* idx, int C1,
                    const float* b, int C2, int N) {
    const unsigned n = (unsigned)N * (unsigned)(C1 + C2);
    k_concat<<<dim3((n + 255) / 256), 256, 0, stream>>>(out, a, idx, C1, b, C2, N);
  };

  // ---- stage 2: graph U-Net ----
  tconv(xb, cN0, 32, e0, cE0, 25, nullptr, nullptr, f1);   // conv0 (no LN), f0 -> f1 buffer
  dlayer(f1, cN0, 32, e0, cE0, 0);
  dlayer(f1, cN0, 32, e0, cE0, 1);
  gather(f2, f1, dn0, cN1, 32);
  dlayer(f2, cN1, 32, e1, cE1, 2);
  dlayer(f2, cN1, 32, e1, cE1, 3);
  gather(f3, f2, dn1, cN2, 32);
  dlayer(f3, cN2, 32, e2, cE2, 4);
  dlayer(f3, cN2, 32, e2, cE2, 5);
  gather(f4, f3, dn2, cN3, 32);
  dlayer(f4, cN3, 32, e3, cE3, 6);
  dlayer(f4, cN3, 32, e3, cE3, 7);
  concat(f5, f4, up2, 32, f3, 32, cN2);
  dlayer(f5, cN2, 64, e2, cE2, 8);
  dlayer(f5, cN2, 64, e2, cE2, 9);
  concat(f6, f5, up1, 64, f2, 32, cN1);
  dlayer(f6, cN1, 96, e1, cE1, 10);
  dlayer(f6, cN1, 96, e1, cE1, 11);
  concat(f7, f6, up0, 96, f1, 32, cN0);
  dlayer(f7, cN0, 128, e0, cE0, 12);
  dlayer(f7, cN0, 128, e0, cE0, 13);

  // ---- stage 3: output head ----
  k_final_head<<<dim3(cN0 / 8), 256, 0, stream>>>(f7, F(173), F(174), F(175), F(176), (float*)d_out, cN0);
}
